// BotRGCN_79448305042029
// MI455X (gfx1250) — matrix operand order repaired and checked by reading, not compiled, not run on hardware
//
#include <hip/hip_runtime.h>
#include <hip/hip_bf16.h>

typedef __attribute__((ext_vector_type(16))) __bf16 v16bf;
typedef __attribute__((ext_vector_type(8)))  float  v8f;

#define KMAX 768   // largest K used by the generic GEMM (des/tweet encoders)

__device__ __forceinline__ float lrelu_f(float v) { return v >= 0.f ? v : 0.01f * v; }

__device__ __forceinline__ v16bf cvt_a_frag(float4 p0, float4 p1, float4 p2, float4 p3)
{
    v16bf a;
    a[0]=(__bf16)p0.x;  a[1]=(__bf16)p0.y;  a[2]=(__bf16)p0.z;  a[3]=(__bf16)p0.w;
    a[4]=(__bf16)p1.x;  a[5]=(__bf16)p1.y;  a[6]=(__bf16)p1.z;  a[7]=(__bf16)p1.w;
    a[8]=(__bf16)p2.x;  a[9]=(__bf16)p2.y;  a[10]=(__bf16)p2.z; a[11]=(__bf16)p2.w;
    a[12]=(__bf16)p3.x; a[13]=(__bf16)p3.y; a[14]=(__bf16)p3.z; a[15]=(__bf16)p3.w;
    return a;
}

// ---------------------------------------------------------------------------
// Generic WMMA GEMM: C[m, col_off+n] = act(A[m,0:K] @ W[0:K,0:Ncols] + bias[n])
// Block = 256 threads = 8 waves, each wave owns a 16x16 tile (rows stacked).
// The block's 16-column weight panel is staged in LDS, transposed, as bf16.
// ---------------------------------------------------------------------------
__global__ void wmma_gemm_kernel(const float* __restrict__ A, int lda,
                                 const float* __restrict__ W,
                                 const float* __restrict__ bias,
                                 float* __restrict__ C, int ldc, int col_off,
                                 int Mrows, int Kdim, int Ncols, int act)
{
    __shared__ __bf16 ldsB[16 * KMAX];
    const int tid  = threadIdx.x;
    const int lane = tid & 31;
    const int wave = tid >> 5;
    const int n0   = blockIdx.y * 16;
    const int Kpad = (Kdim + 31) & ~31;

    // Cooperative fill: LDS[n][k] = bf16(W[k][n0+n]); zero-pad K tail.
    for (int idx = tid; idx < (Kpad << 4); idx += 256) {
        const int k = idx >> 4, n = idx & 15;
        const float f = (k < Kdim) ? W[(long)k * Ncols + n0 + n] : 0.f;
        ldsB[n * Kpad + k] = (__bf16)f;
    }
    __syncthreads();

    const int m0 = blockIdx.x * 128 + wave * 16;
    if (m0 >= Mrows) return;                       // wave-uniform: EXEC stays full
    int row = m0 + (lane & 15);
    if (row >= Mrows) row = Mrows - 1;             // clamp; extra rows masked at store
    const float*  Ap = A + (long)row * lda;
    const int kA = (lane >> 4) * 16;               // lanes 0-15: K 0..15;      16-31: 16..31
    const int kB = (lane >> 4) * 16;               // lanes 0-15: K 0..15;      16-31: 16..31
    const __bf16* Bp = ldsB + (lane & 15) * Kpad;

    v8f acc = {0.f,0.f,0.f,0.f,0.f,0.f,0.f,0.f};
    const int Kfast = Kdim & ~31;
    int k0 = 0;
    for (; k0 < Kfast; k0 += 32) {                 // unguarded, vectorized fast path
        const float4 p0 = *(const float4*)(Ap + k0 + kA);
        const float4 p1 = *(const float4*)(Ap + k0 + kA + 4);
        const float4 p2 = *(const float4*)(Ap + k0 + kA + 8);
        const float4 p3 = *(const float4*)(Ap + k0 + kA + 12);
        const v16bf a = cvt_a_frag(p0, p1, p2, p3);
        const v16bf b = *(const v16bf*)(Bp + k0 + kB);
        acc = __builtin_amdgcn_wmma_f32_16x16x32_bf16(false, a, false, b,
                                                      (short)0, acc, false, false);
    }
    if (k0 < Kdim) {                               // K tail (prop: K=14): clamped loads + selects
        v16bf a;
#pragma unroll
        for (int i = 0; i < 8; ++i) {
            const int k1 = k0 + kA + i;
            const int k2 = k0 + kA + 8 + i;
            const float f1 = Ap[k1 < Kdim ? k1 : 0];
            const float f2 = Ap[k2 < Kdim ? k2 : 0];
            a[i]     = (__bf16)(k1 < Kdim ? f1 : 0.f);
            a[8 + i] = (__bf16)(k2 < Kdim ? f2 : 0.f);
        }
        const v16bf b = *(const v16bf*)(Bp + k0 + kB);   // LDS already zero-padded
        acc = __builtin_amdgcn_wmma_f32_16x16x32_bf16(false, a, false, b,
                                                      (short)0, acc, false, false);
    }

    const float bv = bias ? bias[n0 + (lane & 15)] : 0.f;
#pragma unroll
    for (int r = 0; r < 8; ++r) {
        const int mrow = m0 + ((lane < 16) ? r : 8 + r);
        if (mrow < Mrows) {
            float v = acc[r] + bv;
            if (act) v = lrelu_f(v);
            C[(long)mrow * ldc + col_off + n0 + (lane & 15)] = v;
        }
    }
}

// ---------------------------------------------------------------------------
// RGCN node update: out[n,:] = x[n,:]@W_root + b + sum_r mean_r[n,:]@W_rel[r]
// K=384 virtual concat [x|mean0|mean1]; 1/max(cnt,1) folded into A conversion.
// Weight panel [384 x 16] staged transposed in LDS; 12 WMMAs fully unrolled.
// ---------------------------------------------------------------------------
__global__ void rgcn_node_kernel(const float* __restrict__ x,
                                 const float* __restrict__ summed, // [N,2,128]
                                 const float* __restrict__ cnt,    // [N,2]
                                 const float* __restrict__ Wroot,  // [128,128]
                                 const float* __restrict__ Wrel,   // [2,128,128]
                                 const float* __restrict__ bias,   // [128]
                                 float* __restrict__ out, int Mrows)
{
    __shared__ __bf16 ldsB[16 * 384];
    const int tid  = threadIdx.x;
    const int lane = tid & 31;
    const int wave = tid >> 5;
    const int n0   = blockIdx.y * 16;

    for (int idx = tid; idx < 384 * 16; idx += 256) {
        const int k = idx >> 4, n = idx & 15;
        const float* wrow = (k < 128)
            ? (Wroot + (long)k * 128)
            : (Wrel + ((long)((k >> 7) - 1) * 128 + (k & 127)) * 128);
        ldsB[n * 384 + k] = (__bf16)wrow[n0 + n];
    }
    __syncthreads();

    const int m0 = blockIdx.x * 128 + wave * 16;
    if (m0 >= Mrows) return;
    int row = m0 + (lane & 15);
    if (row >= Mrows) row = Mrows - 1;
    const long r2 = (long)row;
    const int kA = (lane >> 4) * 16;
    const int kB = (lane >> 4) * 16;
    const __bf16* Bp = ldsB + (lane & 15) * 384;

    const float inv0 = 1.f / fmaxf(cnt[r2 * 2 + 0], 1.f);
    const float inv1 = 1.f / fmaxf(cnt[r2 * 2 + 1], 1.f);
    const float* asrc[3] = { x + r2 * 128, summed + r2 * 256, summed + r2 * 256 + 128 };
    const float  ascl[3] = { 1.f, inv0, inv1 };

    v8f acc = {0.f,0.f,0.f,0.f,0.f,0.f,0.f,0.f};
#pragma unroll
    for (int seg = 0; seg < 3; ++seg) {
        const float* Ap = asrc[seg];
        const float  s  = ascl[seg];
#pragma unroll
        for (int kk = 0; kk < 128; kk += 32) {
            float4 p0 = *(const float4*)(Ap + kk + kA);
            float4 p1 = *(const float4*)(Ap + kk + kA + 4);
            float4 p2 = *(const float4*)(Ap + kk + kA + 8);
            float4 p3 = *(const float4*)(Ap + kk + kA + 12);
            p0.x*=s; p0.y*=s; p0.z*=s; p0.w*=s;
            p1.x*=s; p1.y*=s; p1.z*=s; p1.w*=s;
            p2.x*=s; p2.y*=s; p2.z*=s; p2.w*=s;
            p3.x*=s; p3.y*=s; p3.z*=s; p3.w*=s;
            const v16bf a = cvt_a_frag(p0, p1, p2, p3);
            const v16bf b = *(const v16bf*)(Bp + seg * 128 + kk + kB);
            acc = __builtin_amdgcn_wmma_f32_16x16x32_bf16(false, a, false, b,
                                                          (short)0, acc, false, false);
        }
    }

    const float bv = bias[n0 + (lane & 15)];
#pragma unroll
    for (int r = 0; r < 8; ++r) {
        const int mrow = m0 + ((lane < 16) ? r : 8 + r);
        if (mrow < Mrows)
            out[(long)mrow * 128 + n0 + (lane & 15)] = acc[r] + bv;  // no activation
    }
}

// ---------------------------------------------------------------------------
// Edge aggregation: one wave per edge; lane l handles features [4l, 4l+4).
// x (51MB) and summed (102MB) are L2-resident (192MB L2) -> L2 atomic units.
// ---------------------------------------------------------------------------
__global__ void edge_agg_kernel(const float* __restrict__ x,
                                const int* __restrict__ eidx, // [2,E]
                                const int* __restrict__ etyp, // [E]
                                float* __restrict__ summed,
                                float* __restrict__ cnt,
                                long E)
{
    const long gtid = (long)blockIdx.x * blockDim.x + threadIdx.x;
    const long e    = gtid >> 5;
    const int lane  = threadIdx.x & 31;
    if (e >= E) return;
    const int src = eidx[e];
    const int dst = eidx[E + e];
    const int rel = etyp[e];
    const long seg = (long)dst * 2 + rel;
    const float4 v = *(const float4*)(x + (long)src * 128 + lane * 4);
    float* p = summed + seg * 128 + lane * 4;
    atomicAdd(p + 0, v.x);
    atomicAdd(p + 1, v.y);
    atomicAdd(p + 2, v.z);
    atomicAdd(p + 3, v.w);
    if (lane == 0) atomicAdd(cnt + seg, 1.f);
}

__global__ void zero_kernel(float* __restrict__ p, long n)
{
    long i = (long)blockIdx.x * blockDim.x + threadIdx.x;
    const long stride = (long)gridDim.x * blockDim.x;
    for (; i < n; i += stride) p[i] = 0.f;
}

// ---------------------------------------------------------------------------
// Head: logits = h @ W_o2 + b_o2 (128->2), softmax over the 2 classes.
// ---------------------------------------------------------------------------
__global__ void head_kernel(const float* __restrict__ h,
                            const float* __restrict__ Wo2, // [128,2]
                            const float* __restrict__ bo2, // [2]
                            float* __restrict__ out, int Mrows)
{
    const int n = blockIdx.x * blockDim.x + threadIdx.x;
    if (n >= Mrows) return;
    const float* hr = h + (long)n * 128;
    float l0 = bo2[0], l1 = bo2[1];
#pragma unroll 4
    for (int k = 0; k < 128; ++k) {
        const float v = hr[k];
        l0 += v * Wo2[k * 2 + 0];
        l1 += v * Wo2[k * 2 + 1];
    }
    const float m  = fmaxf(l0, l1);
    const float e0 = __expf(l0 - m);
    const float e1 = __expf(l1 - m);
    const float s  = e0 + e1;
    out[(long)n * 2 + 0] = e0 / s;
    out[(long)n * 2 + 1] = e1 / s;
}

extern "C" void kernel_launch(void* const* d_in, const int* in_sizes, int n_in,
                              void* d_out, int out_size, void* d_ws, size_t ws_size,
                              hipStream_t stream)
{
    (void)n_in; (void)out_size; (void)ws_size;
    const float* des    = (const float*)d_in[0];
    const float* tweet  = (const float*)d_in[1];
    const float* prop   = (const float*)d_in[2];
    const int*   eidx   = (const int*)d_in[3];
    const int*   etyp   = (const int*)d_in[4];
    const float* W_des  = (const float*)d_in[5];
    const float* b_des  = (const float*)d_in[6];
    const float* W_tw   = (const float*)d_in[7];
    const float* b_tw   = (const float*)d_in[8];
    const float* W_pr   = (const float*)d_in[9];
    const float* b_pr   = (const float*)d_in[10];
    const float* W_in   = (const float*)d_in[11];
    const float* b_in   = (const float*)d_in[12];
    const float* W_rel  = (const float*)d_in[13];
    const float* W_root = (const float*)d_in[14];
    const float* b_rg   = (const float*)d_in[15];
    const float* W_o1   = (const float*)d_in[16];
    const float* b_o1   = (const float*)d_in[17];
    const float* W_o2   = (const float*)d_in[18];
    const float* b_o2   = (const float*)d_in[19];

    const int  Nn = in_sizes[0] / 768;   // 100000 nodes
    const long E  = in_sizes[4];         // 3.2M edges

    auto aup = [](size_t v) { return (v + 255) & ~(size_t)255; };
    char* w = (char*)d_ws;
    size_t off = 0;
    float* x0     = (float*)(w + off); off += aup((size_t)Nn * 96  * 4);
    float* x1     = (float*)(w + off); off += aup((size_t)Nn * 128 * 4);
    float* summed = (float*)(w + off); off += aup((size_t)Nn * 2 * 128 * 4 +
                                                  (size_t)Nn * 2 * 4);
    float* cntb   = summed + (size_t)Nn * 2 * 128;
    float* x2     = (float*)(w + off); off += aup((size_t)Nn * 128 * 4);
    float* x3     = x1;   // x1 dead after layer-1 node kernel
    float* hbuf   = x2;   // x2 dead after layer-2 node kernel

    const dim3 blk(256);
    const dim3 gQ((Nn + 127) / 128, 2);   // Ncols = 32
    const dim3 gD((Nn + 127) / 128, 8);   // Ncols = 128
    const long zn = (long)Nn * 2 * 128 + (long)Nn * 2;
    const unsigned eblocks = (unsigned)((E * 32 + 255) / 256);

    // Encoders -> x0 = [lrelu(des@Wd+b) | lrelu(tweet@Wt+b) | lrelu(prop@Wp+b)]
    wmma_gemm_kernel<<<gQ, blk, 0, stream>>>(des,   768, W_des, b_des, x0, 96, 0,  Nn, 768, 32, 1);
    wmma_gemm_kernel<<<gQ, blk, 0, stream>>>(tweet, 768, W_tw,  b_tw,  x0, 96, 32, Nn, 768, 32, 1);
    wmma_gemm_kernel<<<gQ, blk, 0, stream>>>(prop,  14,  W_pr,  b_pr,  x0, 96, 64, Nn, 14,  32, 1);
    // Fuse -> x1 = lrelu(x0 @ W_in + b_in)
    wmma_gemm_kernel<<<gD, blk, 0, stream>>>(x0, 96, W_in, b_in, x1, 128, 0, Nn, 96, 128, 1);

    // RGCN layer 1
    zero_kernel<<<1024, blk, 0, stream>>>(summed, zn);
    edge_agg_kernel<<<eblocks, blk, 0, stream>>>(x1, eidx, etyp, summed, cntb, E);
    rgcn_node_kernel<<<gD, blk, 0, stream>>>(x1, summed, cntb, W_root, W_rel, b_rg, x2, Nn);

    // RGCN layer 2
    zero_kernel<<<1024, blk, 0, stream>>>(summed, zn);
    edge_agg_kernel<<<eblocks, blk, 0, stream>>>(x2, eidx, etyp, summed, cntb, E);
    rgcn_node_kernel<<<gD, blk, 0, stream>>>(x2, summed, cntb, W_root, W_rel, b_rg, x3, Nn);

    // Head
    wmma_gemm_kernel<<<gD, blk, 0, stream>>>(x3, 128, W_o1, b_o1, hbuf, 128, 0, Nn, 128, 128, 1);
    head_kernel<<<(Nn + 255) / 256, blk, 0, stream>>>(hbuf, W_o2, b_o2, (float*)d_out, Nn);
}
